// SwinBlock_56676388438020
// MI455X (gfx1250) — matrix operand order repaired and checked by reading, not compiled, not run on hardware
//
#include <hip/hip_runtime.h>
#include <hip/hip_bf16.h>
#include <cstdint>

// ---------------------------------------------------------------------------
// Swin block, MI455X (gfx1250, wave32, WMMA 16x16x32 f16->f32).
// Memory-bound problem => f16 intermediates, fused epilogues, direct-from-
// global WMMA fragment loads (working set is L2-resident: 192MB L2).
// ---------------------------------------------------------------------------

typedef _Float16 halfv16 __attribute__((ext_vector_type(16)));
typedef _Float16 halfv8  __attribute__((ext_vector_type(8)));
typedef float    floatv8 __attribute__((ext_vector_type(8)));

#define MROWS 200704   // 4096 windows * 49 tokens
#define NWIN  4096

__device__ __forceinline__ floatv8 wmma16x16x32(halfv16 a, halfv16 b, floatv8 c) {
  // D = A(16x32) * B(32x16) + C ; emits v_wmma_f32_16x16x32_f16
  return __builtin_amdgcn_wmma_f32_16x16x32_f16(false, a, false, b, (short)0, c, false, false);
}

// A-fragment (16x32 f16, row-major source, contiguous K per lane):
// lane<16 : row=lane,   K = k0..k0+15
// lane>=16: row=lane-16,K = k0+16..k0+31
__device__ __forceinline__ halfv16 load_afrag(const _Float16* __restrict__ A, int ld, int m0, int k0) {
  int lane = threadIdx.x & 31;
  int r = lane & 15;
  int ksel = (lane >> 4) << 4;        // 0 or 16
  const _Float16* p = A + (size_t)(m0 + r) * ld + k0 + ksel;
  halfv8 lo = *(const halfv8*)p;
  halfv8 hi = *(const halfv8*)(p + 8);
  halfv16 out;
#pragma unroll
  for (int i = 0; i < 8; ++i) { out[i] = lo[i]; out[8 + i] = hi[i]; }
  return out;
}

// B-fragment (32x16). Source is B^T (N x K row-major), contiguous per lane:
// lane<16 : col=lane,    K = k0+0..15 ; lane>=16: col=lane-16, K = k0+16..31
__device__ __forceinline__ halfv16 load_bfrag(const _Float16* __restrict__ Bt, int ld, int n0, int k0) {
  int lane = threadIdx.x & 31;
  int n = lane & 15;
  int ksel = (lane >> 4) << 4;        // 0 or 16
  const _Float16* p = Bt + (size_t)(n0 + n) * ld + k0 + ksel;
  return *(const halfv16*)p;          // 32B -> 2x global_load_b128 / ds_load_b128
}

// ---------------------------------------------------------------------------
// Weight transpose f32 (MxN) -> f16 (NxM)
// ---------------------------------------------------------------------------
__global__ void __launch_bounds__(256)
transpose_to_h(const float* __restrict__ src, _Float16* __restrict__ dst, int M, int N) {
  int idx = blockIdx.x * 256 + threadIdx.x;
  if (idx < M * N) {
    int m = idx / N, n = idx - m * N;
    dst[(size_t)n * M + m] = (_Float16)src[idx];
  }
}

// ---------------------------------------------------------------------------
// Row LayerNorm (256 wide), one wave per row (wave32, 8 elems/lane).
// gather=1: row is (window*49+token); read rolled x (roll -3,-3 + partition).
// ---------------------------------------------------------------------------
__global__ void __launch_bounds__(256)
ln_rows(const float* __restrict__ src0, const float* __restrict__ gamma,
        const float* __restrict__ beta, _Float16* __restrict__ out, int gather) {
  int wv = threadIdx.x >> 5, lane = threadIdx.x & 31;
  int row = blockIdx.x * 8 + wv;
  const float* src;
  if (gather) {
    int window = row / 49, token = row - window * 49;
    int bimg = window >> 8, widx = window & 255;
    int ty = token / 7, tx = token - ty * 7;
    int hr = (widx >> 4) * 7 + ty;
    int wr = (widx & 15) * 7 + tx;
    int hs = hr + 3; if (hs >= 112) hs -= 112;
    int wsv = wr + 3; if (wsv >= 112) wsv -= 112;
    src = src0 + ((size_t)bimg * 12544 + hs * 112 + wsv) * 256;
  } else {
    src = src0 + (size_t)row * 256;
  }
  float v[8], s = 0.f, s2 = 0.f;
#pragma unroll
  for (int i = 0; i < 8; ++i) { float t = src[lane + 32 * i]; v[i] = t; s += t; s2 += t * t; }
#pragma unroll
  for (int off = 16; off > 0; off >>= 1) {
    s  += __shfl_xor(s,  off, 32);
    s2 += __shfl_xor(s2, off, 32);
  }
  float mean = s * (1.f / 256.f);
  float var  = s2 * (1.f / 256.f) - mean * mean;
  float rstd = rsqrtf(var + 1e-5f);
#pragma unroll
  for (int i = 0; i < 8; ++i) {
    int c = lane + 32 * i;
    out[(size_t)row * 256 + c] = (_Float16)((v[i] - mean) * rstd * gamma[c] + beta[c]);
  }
}

// ---------------------------------------------------------------------------
// WMMA GEMM: C[M x N] = A[M x K] * Bt[N x K]^T (+ fused epilogue)
// block = 256 thr (8 waves), tile 128(M) x 64(N), wave tile 32x32.
// mode 0: outH = f16(acc + bias)                         (QKV)
// mode 1: outH = f16(gelu(acc + bias))                   (MLP1)
// mode 2: scatter window row -> (b,l); outF = acc+bias+shortcut(x)  (proj)
// mode 3: outF = acc + bias + aux[row,col]               (MLP2 + residual)
// ---------------------------------------------------------------------------
__global__ void __launch_bounds__(256)
gemm_wmma(const _Float16* __restrict__ A, const _Float16* __restrict__ Bt,
          const float* __restrict__ bias, int K, int ldOut,
          _Float16* __restrict__ outH, float* __restrict__ outF,
          const float* __restrict__ aux, int mode) {
  int wv = threadIdx.x >> 5;
  int lane = threadIdx.x & 31;
  int m0 = blockIdx.x * 128 + (wv & 3) * 32;
  int n0 = blockIdx.y * 64 + (wv >> 2) * 32;

  floatv8 acc[2][2] = {};
  for (int k0 = 0; k0 < K; k0 += 32) {
    if (k0 + 32 < K)  // keep next A stripe warm (global_prefetch_b8)
      __builtin_prefetch(A + (size_t)m0 * K + k0 + 32, 0, 1);
    halfv16 a0 = load_afrag(A, K, m0, k0);
    halfv16 a1 = load_afrag(A, K, m0 + 16, k0);
    halfv16 b0 = load_bfrag(Bt, K, n0, k0);
    halfv16 b1 = load_bfrag(Bt, K, n0 + 16, k0);
    acc[0][0] = wmma16x16x32(a0, b0, acc[0][0]);
    acc[0][1] = wmma16x16x32(a0, b1, acc[0][1]);
    acc[1][0] = wmma16x16x32(a1, b0, acc[1][0]);
    acc[1][1] = wmma16x16x32(a1, b1, acc[1][1]);
  }

  int colb = lane & 15;
  int rowb = (lane >> 4) << 3;      // C/D layout: VGPR e -> M = e (+8 for hi lanes)
#pragma unroll
  for (int i = 0; i < 2; ++i) {
#pragma unroll
    for (int e = 0; e < 8; ++e) {
      int row = m0 + 16 * i + rowb + e;
      if (mode == 2) {
        // window row -> rolled coords -> rolled-back output pos; shortcut = rolled x
        int window = row / 49, token = row - window * 49;
        int bimg = window >> 8, widx = window & 255;
        int ty = token / 7, tx = token - ty * 7;
        int hr = (widx >> 4) * 7 + ty;
        int wr = (widx & 15) * 7 + tx;
        int ho = hr + 3; if (ho >= 112) ho -= 112;
        int wo = wr + 3; if (wo >= 112) wo -= 112;
        int hs = hr + 6; if (hs >= 112) hs -= 112;
        int wsv = wr + 6; if (wsv >= 112) wsv -= 112;
        size_t ob = ((size_t)bimg * 12544 + ho * 112 + wo) * 256;
        size_t sb = ((size_t)bimg * 12544 + hs * 112 + wsv) * 256;
#pragma unroll
        for (int j = 0; j < 2; ++j) {
          int col = n0 + 16 * j + colb;
          outF[ob + col] = acc[i][j][e] + bias[col] + aux[sb + col];
        }
      } else {
#pragma unroll
        for (int j = 0; j < 2; ++j) {
          int col = n0 + 16 * j + colb;
          float v = acc[i][j][e] + bias[col];
          size_t o = (size_t)row * ldOut + col;
          if (mode == 0) {
            outH[o] = (_Float16)v;
          } else if (mode == 1) {
            outH[o] = (_Float16)(0.5f * v * (1.0f + erff(v * 0.70710678f)));  // exact GELU
          } else {  // mode 3
            outF[o] = v + aux[o];
          }
        }
      }
    }
  }
}

// ---------------------------------------------------------------------------
// Windowed attention: one block (4 waves) per (window, head).
// S = q k^T (WMMA), softmax(scale*S + relpos + mask) in VALU, O = P V (WMMA).
// 49 tokens padded to 64; head_dim = 32.
// ---------------------------------------------------------------------------
__global__ void __launch_bounds__(128)
attn_win(const _Float16* __restrict__ qkv, const float* __restrict__ relb,
         const int* __restrict__ relidx, const float* __restrict__ mask,
         _Float16* __restrict__ out) {
  __shared__ _Float16 sQ[64 * 32];
  __shared__ _Float16 sK[64 * 32];
  __shared__ _Float16 sVT[32 * 64];
  __shared__ float    sS[64 * 64];
  __shared__ _Float16 sP[64 * 64];

  int tid = threadIdx.x, lane = tid & 31, wv = tid >> 5;
  int window = blockIdx.x >> 3, head = blockIdx.x & 7;
  const _Float16* base = qkv + (size_t)window * 49 * 768 + head * 32;

  // stage q, k (row-major, padded) — 8-half vector chunks
  for (int idx = tid; idx < 64 * 4; idx += 128) {
    int t = idx >> 2, c = (idx & 3) * 8;
    halfv8 zq = {0, 0, 0, 0, 0, 0, 0, 0}, zk = zq;
    if (t < 49) {
      const _Float16* p = base + (size_t)t * 768 + c;
      zq = *(const halfv8*)p;
      zk = *(const halfv8*)(p + 256);
    }
    *(halfv8*)(sQ + t * 32 + c) = zq;
    *(halfv8*)(sK + t * 32 + c) = zk;
  }
  // stage v transposed (32 x 64) so P@V B-fragments are contiguous
  for (int idx = tid; idx < 64 * 32; idx += 128) {
    int t = idx >> 5, d = idx & 31;
    _Float16 v = (_Float16)0.f;
    if (t < 49) v = base[(size_t)t * 768 + 512 + d];
    sVT[d * 64 + t] = v;
  }
  __syncthreads();

  // S = q @ k^T : each wave owns a 16-row stripe, 4 column tiles
  {
    int m0 = wv * 16;
    halfv16 a = load_afrag(sQ, 32, m0, 0);
    int colb = lane & 15, rowb = (lane >> 4) << 3;
#pragma unroll
    for (int j = 0; j < 4; ++j) {
      floatv8 c = {};
      halfv16 b = load_bfrag(sK, 32, j * 16, 0);
      c = wmma16x16x32(a, b, c);
#pragma unroll
      for (int e = 0; e < 8; ++e)
        sS[(m0 + rowb + e) * 64 + j * 16 + colb] = c[e];
    }
  }
  __syncthreads();

  // softmax over valid 49 cols with scale + rel-pos bias + shift mask
  if (tid < 49) {
    int row = tid;
    const int* ri = relidx + row * 49;
    const float* mr = mask + ((size_t)(window & 255) * 49 + row) * 49;
    float mx = -1e30f;
    for (int m = 0; m < 49; ++m) {
      float v = sS[row * 64 + m] * 0.17677669529663687f + relb[ri[m] * 8 + head] + mr[m];
      sS[row * 64 + m] = v;
      mx = fmaxf(mx, v);
    }
    float sum = 0.f;
    for (int m = 0; m < 49; ++m) {
      float e = __expf(sS[row * 64 + m] - mx);
      sS[row * 64 + m] = e;
      sum += e;
    }
    float inv = 1.f / sum;
    for (int m = 0; m < 64; ++m)
      sP[row * 64 + m] = (m < 49) ? (_Float16)(sS[row * 64 + m] * inv) : (_Float16)0.f;
  }
  for (int idx = tid; idx < 15 * 64; idx += 128)      // zero pad rows 49..63
    sP[(49 + (idx >> 6)) * 64 + (idx & 63)] = (_Float16)0.f;
  __syncthreads();

  // O = P @ V  (K = 64 padded, N = 32)
  {
    int m0 = wv * 16;
    floatv8 acc[2] = {};
#pragma unroll
    for (int ks = 0; ks < 64; ks += 32) {
      halfv16 a = load_afrag(sP, 64, m0, ks);
#pragma unroll
      for (int j = 0; j < 2; ++j) {
        halfv16 b = load_bfrag(sVT, 64, j * 16, ks);
        acc[j] = wmma16x16x32(a, b, acc[j]);
      }
    }
    int colb = lane & 15, rowb = (lane >> 4) << 3;
#pragma unroll
    for (int j = 0; j < 2; ++j) {
#pragma unroll
      for (int e = 0; e < 8; ++e) {
        int row = m0 + rowb + e;
        if (row < 49)
          out[((size_t)window * 49 + row) * 256 + head * 32 + j * 16 + colb] = (_Float16)acc[j][e];
      }
    }
  }
}

// ---------------------------------------------------------------------------
extern "C" void kernel_launch(void* const* d_in, const int* in_sizes, int n_in,
                              void* d_out, int out_size, void* d_ws, size_t ws_size,
                              hipStream_t stream) {
  (void)in_sizes; (void)n_in; (void)out_size; (void)ws_size;
  const float* x      = (const float*)d_in[0];
  const float* qkv_w  = (const float*)d_in[1];
  const float* qkv_b  = (const float*)d_in[2];
  const float* proj_w = (const float*)d_in[3];
  const float* proj_b = (const float*)d_in[4];
  const float* relb   = (const float*)d_in[5];
  const float* n1s    = (const float*)d_in[6];
  const float* n1b    = (const float*)d_in[7];
  const float* n2s    = (const float*)d_in[8];
  const float* n2b    = (const float*)d_in[9];
  const float* w1     = (const float*)d_in[10];
  const float* b1     = (const float*)d_in[11];
  const float* w2     = (const float*)d_in[12];
  const float* b2     = (const float*)d_in[13];
  const int*   relidx = (const int*)d_in[14];
  const float* amask  = (const float*)d_in[15];
  float* out = (float*)d_out;

  const size_t M = MROWS;
  char* w = (char*)d_ws;
  // region 1 (M*1024 f16): holds qkv (M*768) + attnout (M*256) early, hidden later
  _Float16* qkvbuf  = (_Float16*)w;
  _Float16* hidden  = (_Float16*)w;
  _Float16* attnout = qkvbuf + M * 768;
  w += M * 1024 * sizeof(_Float16);
  // region 2 (M*256 f16): xn (LN1 out) early, h_in (LN2 out) later
  _Float16* xn  = (_Float16*)w;
  _Float16* hin = (_Float16*)w;
  w += M * 256 * sizeof(_Float16);
  // region 3: xr residual (f32)
  float* xr = (float*)w;
  w += M * 256 * sizeof(float);
  // transposed f16 weights
  _Float16* qkvT = (_Float16*)w; w += 768 * 256 * sizeof(_Float16);
  _Float16* projT = (_Float16*)w; w += 256 * 256 * sizeof(_Float16);
  _Float16* w1T  = (_Float16*)w; w += 1024 * 256 * sizeof(_Float16);
  _Float16* w2T  = (_Float16*)w; w += 256 * 1024 * sizeof(_Float16);
  // total workspace ~688 MB

  transpose_to_h<<<(256 * 768 + 255) / 256, 256, 0, stream>>>(qkv_w, qkvT, 256, 768);
  transpose_to_h<<<(256 * 256 + 255) / 256, 256, 0, stream>>>(proj_w, projT, 256, 256);
  transpose_to_h<<<(256 * 1024 + 255) / 256, 256, 0, stream>>>(w1, w1T, 256, 1024);
  transpose_to_h<<<(1024 * 256 + 255) / 256, 256, 0, stream>>>(w2, w2T, 1024, 256);

  // roll + partition + LN1 -> xn (f16)
  ln_rows<<<M / 8, 256, 0, stream>>>(x, n1s, n1b, xn, 1);
  // QKV GEMM (+bias) -> f16
  gemm_wmma<<<dim3(M / 128, 12), 256, 0, stream>>>(xn, qkvT, qkv_b, 256, 768, qkvbuf, nullptr, nullptr, 0);
  // windowed attention
  attn_win<<<NWIN * 8, 128, 0, stream>>>(qkvbuf, relb, relidx, amask, attnout);
  // proj GEMM + reverse/roll scatter + shortcut residual -> xr (f32, (B,L,C))
  gemm_wmma<<<dim3(M / 128, 4), 256, 0, stream>>>(attnout, projT, proj_b, 256, 256, nullptr, xr, x, 2);
  // LN2 -> h_in (f16)
  ln_rows<<<M / 8, 256, 0, stream>>>(xr, n2s, n2b, hin, 0);
  // MLP1 GEMM + GELU -> hidden (f16)
  gemm_wmma<<<dim3(M / 128, 16), 256, 0, stream>>>(hin, w1T, b1, 256, 1024, hidden, nullptr, nullptr, 1);
  // MLP2 GEMM + bias + xr residual -> d_out (f32)
  gemm_wmma<<<dim3(M / 128, 4), 256, 0, stream>>>(hidden, w2T, b2, 1024, 256, nullptr, out, xr, 3);
}
